// LocalSubgraphEncoder_82918638617235
// MI455X (gfx1250) — compile-verified, matrix operand order checked by reading, not run on hardware
//
#include <hip/hip_runtime.h>
#include <hip/hip_bf16.h>
#include <stdint.h>

#define N_NODES 4096
#define DMODEL  256
#define NHEAD   8
#define DHEAD   32
#define NEDGE   131072
#define NTYPES  16
#define LN_EPS  1e-5f

typedef __bf16 bf16_t;
typedef bf16_t bf16x8  __attribute__((ext_vector_type(8)));
typedef bf16_t bf16x16 __attribute__((ext_vector_type(16)));
typedef float  f32x8   __attribute__((ext_vector_type(8)));
typedef unsigned int u32x4 __attribute__((ext_vector_type(4)));
typedef int    i32x8   __attribute__((ext_vector_type(8)));
typedef int    i32x4   __attribute__((ext_vector_type(4)));

// ---------------------------------------------------------------- WMMA helper
__device__ __forceinline__ f32x8 wmma_bf16(bf16x16 a, bf16x16 b, f32x8 c) {
  return __builtin_amdgcn_wmma_f32_16x16x32_bf16(false, a, false, b, (short)0, c,
                                                 false, false);
}

// ------------------------------------------- CDNA5 async global->LDS staging
__device__ __forceinline__ uint32_t lds_off_of(const void* p) {
  // generic pointers to LDS on amdgcn: low 32 bits == LDS offset
  return (uint32_t)(uintptr_t)p;
}
__device__ __forceinline__ void async_copy_b128(uint32_t lds_off, const void* gaddr) {
  asm volatile("global_load_async_to_lds_b128 %0, %1, off"
               :: "v"(lds_off), "v"(gaddr) : "memory");
}
__device__ __forceinline__ void wait_async0() {
  asm volatile("s_wait_asynccnt 0" ::: "memory");
}

// --------------------------------------------------- Tensor Data Mover (TDM)
// 2D tile load: tile_w elements (2B each) x tile_h rows, row stride in elems.
// D# packing per CDNA5 ISA.
// This toolchain exposes the 6-arg builtin:
//   (u32x4 g0, i32x8 g1, i32x4 g2, i32x4 g3, i32x8 pad, i32 cpol)
__device__ __forceinline__ void tdm_load_2d(uint32_t lds_addr, const void* gaddr,
                                            uint32_t tile_w, uint32_t tile_h,
                                            uint32_t stride_elems) {
  uint64_t ga = (uint64_t)(uintptr_t)gaddr;
  u32x4 g0;
  g0[0] = 1u;                                    // count=1, user descriptor
  g0[1] = lds_addr;                              // lds_addr [63:32]
  g0[2] = (uint32_t)ga;                          // global_addr lo
  g0[3] = (uint32_t)(ga >> 32) | (2u << 30);     // global_addr hi | type=2
  i32x8 g1;
  g1[0] = (int)(1u << 16);                       // data_size=1 (2 bytes)
  g1[1] = (int)((tile_w & 0xFFFFu) << 16);       // tensor_dim0[15:0]
  g1[2] = (int)(((tile_w >> 16) & 0xFFFFu) |     // tensor_dim0[31:16]
                ((tile_h & 0xFFFFu) << 16));     // tensor_dim1[15:0]
  g1[3] = (int)(((tile_h >> 16) & 0xFFFFu) |     // tensor_dim1[31:16]
                ((tile_w & 0xFFFFu) << 16));     // tile_dim0
  g1[4] = (int)(tile_h & 0xFFFFu);               // tile_dim1 (tile_dim2=0)
  g1[5] = (int)stride_elems;                     // tensor_dim0_stride[31:0]
  g1[6] = 0;                                     // stride hi, dim1_stride lo
  g1[7] = 0;
  i32x4 z4 = {};                                 // groups 2/3 unused (2D)
  i32x8 z8 = {};
  __builtin_amdgcn_tensor_load_to_lds(g0, g1, z4, z4, z8, 0);
}

// ------------------------------------------------------------ fragment loads
// 16-bit A/B fragment: lane l -> row (l&15), chunks at k=(l>>4)*8 and +16,
// each chunk is 8 contiguous bf16 (16B) in a row of length `stride` elems.
__device__ __forceinline__ bf16x16 load_frag(const bf16_t* base, int stride) {
  int lane = threadIdx.x & 31;
  int r  = lane & 15;
  int kb = (lane >> 4) * 8;
  bf16x8 lo = *reinterpret_cast<const bf16x8*>(base + (size_t)r * stride + kb);
  bf16x8 hi = *reinterpret_cast<const bf16x8*>(base + (size_t)r * stride + kb + 16);
  return __builtin_shufflevector(lo, hi, 0,1,2,3,4,5,6,7,8,9,10,11,12,13,14,15);
}

// =============================================================== prep kernel
// hB = bf16(x + pos); WT[n][k] = bf16(W[k][n]) for the four weight matrices.
__global__ void prep_kernel(const float* __restrict__ x, const float* __restrict__ pos,
                            const float* __restrict__ Wq, const float* __restrict__ Wk,
                            const float* __restrict__ Wv, const float* __restrict__ Wo,
                            bf16_t* hB, bf16_t* WqT, bf16_t* WkT, bf16_t* WvT, bf16_t* WoT) {
  int i = blockIdx.x * blockDim.x + threadIdx.x;
  const int HN = N_NODES * DMODEL;
  if (i < HN) { hB[i] = (bf16_t)(x[i] + pos[i]); return; }
  int j = i - HN;
  if (j >= 4 * DMODEL * DMODEL) return;
  int w = j >> 16;           // which matrix (65536 elems each)
  int e = j & 65535;
  int k = e >> 8, n = e & 255;
  const float* W = (w == 0) ? Wq : (w == 1) ? Wk : (w == 2) ? Wv : Wo;
  bf16_t*      T = (w == 0) ? WqT : (w == 1) ? WkT : (w == 2) ? WvT : WoT;
  T[n * DMODEL + k] = (bf16_t)W[k * DMODEL + n];
}

// ======================================================= CSR build (3 steps)
__global__ void zero_u32_kernel(uint32_t* p, int n) {
  int i = blockIdx.x * blockDim.x + threadIdx.x;
  if (i < n) p[i] = 0u;
}
__global__ void count_kernel(const int* __restrict__ eidx, uint32_t* cnt) {
  int e = blockIdx.x * blockDim.x + threadIdx.x;
  if (e < NEDGE) atomicAdd(&cnt[eidx[e]], 1u);
}
__global__ __launch_bounds__(1024)
void scan_kernel(const uint32_t* __restrict__ cnt, uint32_t* rowptr, uint32_t* cursor) {
  __shared__ uint32_t s[1024];
  int tid = threadIdx.x;
  int base = tid * 4;
  uint32_t c0 = cnt[base], c1 = cnt[base+1], c2 = cnt[base+2], c3 = cnt[base+3];
  uint32_t tot = c0 + c1 + c2 + c3;
  s[tid] = tot;
  __syncthreads();
  for (int off = 1; off < 1024; off <<= 1) {
    uint32_t v = (tid >= off) ? s[tid - off] : 0u;
    __syncthreads();
    s[tid] += v;
    __syncthreads();
  }
  uint32_t run = s[tid] - tot;          // exclusive prefix
  rowptr[base]   = run; cursor[base]   = run; run += c0;
  rowptr[base+1] = run; cursor[base+1] = run; run += c1;
  rowptr[base+2] = run; cursor[base+2] = run; run += c2;
  rowptr[base+3] = run; cursor[base+3] = run; run += c3;
  if (tid == 1023) rowptr[N_NODES] = run;
}
__global__ void scatter_kernel(const int* __restrict__ eidx, const int* __restrict__ etyp,
                               uint32_t* cursor, uint32_t* ecol, uint32_t* etype_out) {
  int e = blockIdx.x * blockDim.x + threadIdx.x;
  if (e >= NEDGE) return;
  int src = eidx[e];
  int dst = eidx[NEDGE + e];
  uint32_t pos = atomicAdd(&cursor[src], 1u);
  ecol[pos]      = (uint32_t)dst;
  etype_out[pos] = (uint32_t)etyp[e];
}

// ================== bf16 WMMA GEMM (TDM-staged, double-buffered): A*W^T+bias
// Block = 128 threads (4 waves). Tile 64x64, K stepped by 32. Wave 0 drives
// the Tensor Data Mover; compute overlaps the next tile's DMA.
template <bool F32OUT>
__global__ __launch_bounds__(128)
void gemm_kernel(const bf16_t* __restrict__ A, const bf16_t* __restrict__ WT,
                 const float* __restrict__ bias, const float* __restrict__ xres,
                 bf16_t* __restrict__ OutB, float* __restrict__ OutF) {
  __shared__ bf16_t At[2][64][32];
  __shared__ bf16_t Bt[2][64][32];
  int t = threadIdx.x, wv = t >> 5, lane = t & 31;
  int m0 = (blockIdx.x & 63) * 64;
  int n0 = (blockIdx.x >> 6) * 64;
  f32x8 acc[4] = {};

  const int KSTEPS = DMODEL / 32;
  if (wv == 0) {  // prologue: stage tile 0
    tdm_load_2d(lds_off_of(&At[0][0][0]), A  + (size_t)m0 * DMODEL, 32, 64, DMODEL);
    tdm_load_2d(lds_off_of(&Bt[0][0][0]), WT + (size_t)n0 * DMODEL, 32, 64, DMODEL);
  }
  for (int ks = 0; ks < KSTEPS; ++ks) {
    int buf = ks & 1;
    if (wv == 0) {
      if (ks + 1 < KSTEPS) {  // stage tile ks+1 while computing tile ks
        int k1 = (ks + 1) * 32;
        tdm_load_2d(lds_off_of(&At[buf ^ 1][0][0]),
                    A  + (size_t)m0 * DMODEL + k1, 32, 64, DMODEL);
        tdm_load_2d(lds_off_of(&Bt[buf ^ 1][0][0]),
                    WT + (size_t)n0 * DMODEL + k1, 32, 64, DMODEL);
        __builtin_amdgcn_s_wait_tensorcnt(2);   // tile ks landed
      } else {
        __builtin_amdgcn_s_wait_tensorcnt(0);
      }
    }
    __syncthreads();
    bf16x16 af = load_frag(&At[buf][wv * 16][0], 32);
#pragma unroll
    for (int nt = 0; nt < 4; ++nt) {
      bf16x16 bfr = load_frag(&Bt[buf][nt * 16][0], 32);
      acc[nt] = wmma_bf16(af, bfr, acc[nt]);
    }
    __syncthreads();   // everyone done reading before buf is re-staged
  }
  int hi = lane >> 4, lc = lane & 15;
#pragma unroll
  for (int nt = 0; nt < 4; ++nt) {
    int col = n0 + nt * 16 + lc;
    float bc = bias[col];
#pragma unroll
    for (int r = 0; r < 8; ++r) {
      int row = m0 + wv * 16 + r + hi * 8;
      if (F32OUT)
        OutF[(size_t)row * DMODEL + col] =
            acc[nt][r] + bc + xres[(size_t)row * DMODEL + col];
      else
        OutB[(size_t)row * DMODEL + col] = (bf16_t)(acc[nt][r] + bc);
    }
  }
}

// ==================================== flash attention with sparse edge bias
// Block: 128 threads = 4 waves; blockIdx = h*64 + rowblock. Each wave owns 16
// rows and sweeps all 4096 columns in 32-wide tiles (online softmax).
// K tiles: async global->LDS (ASYNCcnt); V tiles: transposed manual staging.
// Double buffered: tile i+1 is staged while tile i is consumed.
__global__ __launch_bounds__(128)
void attn_kernel(const bf16_t* __restrict__ QB, const bf16_t* __restrict__ KB,
                 const bf16_t* __restrict__ VB,
                 const uint32_t* __restrict__ rowptr, const uint32_t* __restrict__ ecol,
                 const uint32_t* __restrict__ etype, const float* __restrict__ edge_emb,
                 bf16_t* __restrict__ attnB) {
  __shared__ bf16_t Kt[2][32][32];         // [n][d], async-staged
  __shared__ bf16_t Vt[2][32][32];         // [d][n], transposed while staging
  __shared__ float  Sw[4][16][32];         // per-wave score tile
  __shared__ bf16_t Pw[4][16][32];         // per-wave softmax tile (bf16)
  __shared__ float  srow[4][16];           // per-row rescale factor
  __shared__ float  lrow[4][16];           // per-row softmax denominator
  __shared__ float  semb[NTYPES * NHEAD];

  int t = threadIdx.x, wv = t >> 5, lane = t & 31;
  int lc = lane & 15, hi = lane >> 4;
  int h  = blockIdx.x >> 6;
  int r0 = (blockIdx.x & 63) * 64;
  int srow_stage = t >> 2, sch = (t & 3) * 8;   // staging role of this thread

  if (t < NTYPES * NHEAD) semb[t] = edge_emb[t];

  // Q fragment for this wave's 16 rows, held in registers for the whole sweep
  bf16x16 qf = load_frag(QB + (size_t)(r0 + wv * 16) * DMODEL + h * DHEAD, DMODEL);
  f32x8 o0 = {}, o1 = {};
  float mrun = -3.0e38f, lrun = 0.0f;
  const float scl = 0.17677669529663687f;  // 1/sqrt(32)

  // prologue: stage tile 0
  {
    async_copy_b128(lds_off_of(&Kt[0][srow_stage][sch]),
                    KB + (size_t)srow_stage * DMODEL + h * DHEAD + sch);
    bf16x8 v = *reinterpret_cast<const bf16x8*>(
        VB + (size_t)srow_stage * DMODEL + h * DHEAD + sch);
#pragma unroll
    for (int j = 0; j < 8; ++j) Vt[0][sch + j][srow_stage] = v[j];
  }
  wait_async0();
  __syncthreads();

  for (int c0 = 0; c0 < N_NODES; c0 += 32) {
    int buf = (c0 >> 5) & 1;
    // ---- stage tile c0+32 into the other buffer (overlaps compute below)
    if (c0 + 32 < N_NODES) {
      int gr = c0 + 32 + srow_stage;
      async_copy_b128(lds_off_of(&Kt[buf ^ 1][srow_stage][sch]),
                      KB + (size_t)gr * DMODEL + h * DHEAD + sch);
      bf16x8 v = *reinterpret_cast<const bf16x8*>(
          VB + (size_t)gr * DMODEL + h * DHEAD + sch);
#pragma unroll
      for (int j = 0; j < 8; ++j) Vt[buf ^ 1][sch + j][srow_stage] = v[j];
    }

    // ---- S = (Q K^T) * scl, spilled to LDS
#pragma unroll
    for (int nt = 0; nt < 2; ++nt) {
      f32x8 z = {};
      bf16x16 bfr = load_frag(&Kt[buf][nt * 16][0], 32);
      f32x8 s = wmma_bf16(qf, bfr, z);
#pragma unroll
      for (int r = 0; r < 8; ++r)
        Sw[wv][r + hi * 8][nt * 16 + lc] = s[r] * scl;
    }
    __syncthreads();

    // ---- sparse bias + online softmax (one lane per row)
    if (lane < 16) {
      int m = lane;
      int g = r0 + wv * 16 + m;
      uint32_t b = rowptr[g], e = rowptr[g + 1];
      for (uint32_t idx = b; idx < e; ++idx) {
        uint32_t c = ecol[idx];
        if (c >= (uint32_t)c0 && c < (uint32_t)(c0 + 32))
          Sw[wv][m][c - c0] += semb[etype[idx] * NHEAD + h];
      }
      float tmax = -3.0e38f;
#pragma unroll
      for (int j = 0; j < 32; ++j) tmax = fmaxf(tmax, Sw[wv][m][j]);
      float nm = fmaxf(mrun, tmax);
      float sf = __expf(mrun - nm);
      float tsum = 0.0f;
#pragma unroll
      for (int j = 0; j < 32; ++j) {
        float p = __expf(Sw[wv][m][j] - nm);
        tsum += p;
        Pw[wv][m][j] = (bf16_t)p;
      }
      lrun = lrun * sf + tsum;
      mrun = nm;
      srow[wv][m] = sf;
    }
    __syncthreads();

    // ---- rescale O and accumulate P*V
#pragma unroll
    for (int r = 0; r < 8; ++r) {
      float sf = srow[wv][r + hi * 8];
      o0[r] *= sf;
      o1[r] *= sf;
    }
    bf16x16 pf = load_frag(&Pw[wv][0][0], 32);
    bf16x16 v0 = load_frag(&Vt[buf][0][0], 32);   // d = 0..15
    bf16x16 v1 = load_frag(&Vt[buf][16][0], 32);  // d = 16..31
    o0 = wmma_bf16(pf, v0, o0);
    o1 = wmma_bf16(pf, v1, o1);

    wait_async0();      // next K tile landed (copy overlapped the compute)
    __syncthreads();    // all waves done with buf before it is re-staged
  }

  if (lane < 16) lrow[wv][lane] = lrun;
  __syncthreads();
#pragma unroll
  for (int r = 0; r < 8; ++r) {
    int m = r + hi * 8;
    float inv = 1.0f / lrow[wv][m];
    int row = r0 + wv * 16 + m;
    attnB[(size_t)row * DMODEL + h * DHEAD + lc]      = (bf16_t)(o0[r] * inv);
    attnB[(size_t)row * DMODEL + h * DHEAD + 16 + lc] = (bf16_t)(o1[r] * inv);
  }
}

// ================================================== LayerNorm (wave per row)
__global__ __launch_bounds__(32)
void ln_kernel(const float* __restrict__ proj, const float* __restrict__ gamma,
               const float* __restrict__ beta, float* __restrict__ out) {
  int row = blockIdx.x, lane = threadIdx.x;
  const float* p = proj + (size_t)row * DMODEL + lane * 8;
  float v[8], s1 = 0.0f, s2 = 0.0f;
#pragma unroll
  for (int j = 0; j < 8; ++j) { v[j] = p[j]; s1 += v[j]; s2 += v[j] * v[j]; }
#pragma unroll
  for (int off = 16; off > 0; off >>= 1) {
    s1 += __shfl_xor(s1, off, 32);
    s2 += __shfl_xor(s2, off, 32);
  }
  float mu  = s1 * (1.0f / DMODEL);
  float var = s2 * (1.0f / DMODEL) - mu * mu;
  float inv = rsqrtf(var + LN_EPS);
  float* o = out + (size_t)row * DMODEL + lane * 8;
#pragma unroll
  for (int j = 0; j < 8; ++j)
    o[j] = (v[j] - mu) * inv * gamma[lane * 8 + j] + beta[lane * 8 + j];
}

// ============================================================== launch glue
extern "C" void kernel_launch(void* const* d_in, const int* in_sizes, int n_in,
                              void* d_out, int out_size, void* d_ws, size_t ws_size,
                              hipStream_t stream) {
  (void)in_sizes; (void)n_in; (void)out_size; (void)ws_size;
  const float* x    = (const float*)d_in[0];
  const float* pos  = (const float*)d_in[1];
  const int*   eidx = (const int*)d_in[2];   // [2, E]
  const int*   etyp = (const int*)d_in[3];   // [E]
  const float* Wq   = (const float*)d_in[4];
  const float* bq   = (const float*)d_in[5];
  const float* Wk   = (const float*)d_in[6];
  const float* bk   = (const float*)d_in[7];
  const float* Wv   = (const float*)d_in[8];
  const float* bv   = (const float*)d_in[9];
  const float* Wo   = (const float*)d_in[10];
  const float* bo   = (const float*)d_in[11];
  const float* eemb = (const float*)d_in[12];
  const float* gam  = (const float*)d_in[13];
  const float* bet  = (const float*)d_in[14];
  float* out = (float*)d_out;

  char* ws = (char*)d_ws;
  size_t off = 0;
  auto take = [&](size_t bytes) -> char* {
    char* p = ws + off;
    off = (off + bytes + 255) & ~(size_t)255;
    return p;
  };
  const size_t ND = (size_t)N_NODES * DMODEL;
  bf16_t*   hB     = (bf16_t*)take(2 * ND);
  bf16_t*   WqT    = (bf16_t*)take(2 * DMODEL * DMODEL);
  bf16_t*   WkT    = (bf16_t*)take(2 * DMODEL * DMODEL);
  bf16_t*   WvT    = (bf16_t*)take(2 * DMODEL * DMODEL);
  bf16_t*   WoT    = (bf16_t*)take(2 * DMODEL * DMODEL);
  bf16_t*   QB     = (bf16_t*)take(2 * ND);
  bf16_t*   KB     = (bf16_t*)take(2 * ND);
  bf16_t*   VB     = (bf16_t*)take(2 * ND);
  bf16_t*   attnB  = (bf16_t*)take(2 * ND);
  float*    proj   = (float*)take(4 * ND);
  uint32_t* cnt    = (uint32_t*)take(4 * N_NODES);
  uint32_t* rowptr = (uint32_t*)take(4 * (N_NODES + 1));
  uint32_t* cursor = (uint32_t*)take(4 * N_NODES);
  uint32_t* ecol   = (uint32_t*)take(4 * NEDGE);
  uint32_t* etype  = (uint32_t*)take(4 * NEDGE);

  // CSR build
  zero_u32_kernel<<<(N_NODES + 255) / 256, 256, 0, stream>>>(cnt, N_NODES);
  count_kernel<<<NEDGE / 256, 256, 0, stream>>>(eidx, cnt);
  scan_kernel<<<1, 1024, 0, stream>>>(cnt, rowptr, cursor);
  scatter_kernel<<<NEDGE / 256, 256, 0, stream>>>(eidx, etyp, cursor, ecol, etype);

  // prep: h = x+pos (bf16) + weight transposes
  {
    int total = N_NODES * DMODEL + 4 * DMODEL * DMODEL;
    prep_kernel<<<(total + 255) / 256, 256, 0, stream>>>(x, pos, Wq, Wk, Wv, Wo,
                                                         hB, WqT, WkT, WvT, WoT);
  }

  // Q/K/V projections (WMMA, TDM-staged)
  dim3 ggrid(64 * (DMODEL / 64));  // 64 M-tiles x 4 N-tiles = 256 blocks
  gemm_kernel<false><<<ggrid, 128, 0, stream>>>(hB, WqT, bq, nullptr, QB, nullptr);
  gemm_kernel<false><<<ggrid, 128, 0, stream>>>(hB, WkT, bk, nullptr, KB, nullptr);
  gemm_kernel<false><<<ggrid, 128, 0, stream>>>(hB, WvT, bv, nullptr, VB, nullptr);

  // flash attention with edge bias
  attn_kernel<<<NHEAD * (N_NODES / 64), 128, 0, stream>>>(
      QB, KB, VB, rowptr, ecol, etype, eemb, attnB);

  // output projection + residual, then LayerNorm
  gemm_kernel<true><<<ggrid, 128, 0, stream>>>(attnB, WoT, bo, x, nullptr, proj);
  ln_kernel<<<N_NODES, 32, 0, stream>>>(proj, gam, bet, out);
}
